// MultiHeadSelfAttention_51539607837
// MI455X (gfx1250) — matrix operand order repaired and checked by reading, not compiled, not run on hardware
//
#include <hip/hip_runtime.h>
#include <hip/hip_bf16.h>

// ---------------------------------------------------------------------------
// MI455X (gfx1250, wave32) multi-head self-attention, f16 WMMA pipeline.
// D = A*B + C via v_wmma_f32_16x16x32_f16 (probe-verified builtin).
// ---------------------------------------------------------------------------

#define NHEAD   12
#define HD      64
#define DMODEL  768
#define SEQ     4096
#define NBATCH  2
#define MTOT    (NBATCH * SEQ)      // 8192 rows
#define LN_EPS  1e-5f
#define NEG_INF -1e9f

typedef __attribute__((ext_vector_type(16))) _Float16 v16h;
typedef __attribute__((ext_vector_type(8)))  _Float16 h8;
typedef __attribute__((ext_vector_type(8)))  float    v8f;

union FragA { h8 h[2]; v16h v; };   // A: two 16B chunks per lane

static __device__ __forceinline__ v8f wmma16(v16h a, v16h b, v8f c) {
    // 8 args: (neg_a, A, neg_b, B, c_mod, C, reuse_a, reuse_b)
    return __builtin_amdgcn_wmma_f32_16x16x32_f16(false, a, false, b,
                                                  (short)0, c, false, false);
}

static __device__ __forceinline__ _Float16 f2h(float f) { return (_Float16)f; }

// ---------------------------------------------------------------------------
// Kernel 0: cast the 4 weight matrices f32 -> f16 (stored [4][D*D])
// ---------------------------------------------------------------------------
__global__ void cvt_w_kernel(const float* __restrict__ s0, const float* __restrict__ s1,
                             const float* __restrict__ s2, const float* __restrict__ s3,
                             _Float16* __restrict__ dst) {
    int z = blockIdx.y;
    const float* s = (z == 0) ? s0 : (z == 1) ? s1 : (z == 2) ? s2 : s3;
    size_t i = (size_t)blockIdx.x * blockDim.x + threadIdx.x;
    dst[(size_t)z * DMODEL * DMODEL + i] = f2h(s[i]);
}

// ---------------------------------------------------------------------------
// Kernel 1: pre-LayerNorm, write xn as f16 [MTOT, D]. One block per row.
// ---------------------------------------------------------------------------
__global__ __launch_bounds__(256) void ln_kernel(const float* __restrict__ x,
                                                 const float* __restrict__ gamma,
                                                 const float* __restrict__ beta,
                                                 _Float16* __restrict__ xn) {
    int row = blockIdx.x;
    int t   = threadIdx.x;                 // 256 threads, 3 elems each
    const float* xr = x + (size_t)row * DMODEL;
    float v0 = xr[t], v1 = xr[t + 256], v2 = xr[t + 512];
    float s  = v0 + v1 + v2;
    float s2 = v0 * v0 + v1 * v1 + v2 * v2;
    for (int m = 16; m; m >>= 1) { s += __shfl_xor(s, m); s2 += __shfl_xor(s2, m); }
    __shared__ float ssum[8], ssq[8];
    int wid = t >> 5;
    if ((t & 31) == 0) { ssum[wid] = s; ssq[wid] = s2; }
    __syncthreads();
    float ts = 0.f, ts2 = 0.f;
    #pragma unroll
    for (int i = 0; i < 8; i++) { ts += ssum[i]; ts2 += ssq[i]; }
    float mu  = ts / DMODEL;
    float var = ts2 / DMODEL - mu * mu;
    float inv = rsqrtf(var + LN_EPS);
    _Float16* xo = xn + (size_t)row * DMODEL;
    xo[t]       = f2h((v0 - mu) * inv * gamma[t]       + beta[t]);
    xo[t + 256] = f2h((v1 - mu) * inv * gamma[t + 256] + beta[t + 256]);
    xo[t + 512] = f2h((v2 - mu) * inv * gamma[t + 512] + beta[t + 512]);
}

// ---------------------------------------------------------------------------
// Kernel 2: fused QKV projection, y = xn @ W^T + b  (NT GEMM, M=8192 N=768 K=768)
// blockDim 128 (4 waves); each wave owns a 32x64 output tile (2 M-subtiles x
// 4 N-subtiles -> 8 independent WMMA chains per K-step for latency hiding;
// each B frag feeds 2 WMMAs). grid = (M/128, N/64, 3) ; z selects Q/K/V.
// Q,K written [B,H,S,HD]; V written transposed [B,H,HD,S] for the PV GEMM.
// ---------------------------------------------------------------------------
__global__ __launch_bounds__(128) void qkv_kernel(const _Float16* __restrict__ xn,
                                                  const _Float16* __restrict__ wh, // [3][D*D]
                                                  const float* __restrict__ bq,
                                                  const float* __restrict__ bk,
                                                  const float* __restrict__ bv,
                                                  _Float16* __restrict__ qo,
                                                  _Float16* __restrict__ ko,
                                                  _Float16* __restrict__ vo) {
    int z = blockIdx.z;
    const _Float16* w   = wh + (size_t)z * DMODEL * DMODEL;
    const float*    bia = (z == 0) ? bq : (z == 1) ? bk : bv;
    int lane = threadIdx.x & 31, wave = threadIdx.x >> 5;
    int lm = lane & 15, lh = lane >> 4;
    int m0 = blockIdx.x * 128 + wave * 32;
    int n0 = blockIdx.y * 64;

    const _Float16* arow0 = xn + (size_t)(m0 + lm) * DMODEL;
    const _Float16* arow1 = xn + (size_t)(m0 + 16 + lm) * DMODEL;
    const _Float16* brow  = w  + (size_t)(n0 + lm) * DMODEL;
    v8f acc[2][4] = {};
    for (int k = 0; k < DMODEL; k += 32) {
        if (k + 64 < DMODEL) {                // pull next K-chunk into WGP$
            __builtin_prefetch(arow0 + k + 64, 0, 0);
            __builtin_prefetch(arow1 + k + 64, 0, 0);
            __builtin_prefetch(brow + k + 64, 0, 0);
        }
        FragA a0, a1;
        a0.h[0] = *(const h8*)(arow0 + k + lh * 16);
        a0.h[1] = *(const h8*)(arow0 + k + 8 + lh * 16);
        a1.h[0] = *(const h8*)(arow1 + k + lh * 16);
        a1.h[1] = *(const h8*)(arow1 + k + 8 + lh * 16);
        #pragma unroll
        for (int nt = 0; nt < 4; nt++) {
            v16h b = *(const v16h*)(w + (size_t)(n0 + nt * 16 + lm) * DMODEL + k + lh * 16);
            acc[0][nt] = wmma16(a0.v, b, acc[0][nt]);
            acc[1][nt] = wmma16(a1.v, b, acc[1][nt]);
        }
    }
    #pragma unroll
    for (int mt = 0; mt < 2; mt++) {
        #pragma unroll
        for (int nt = 0; nt < 4; nt++) {
            int n = n0 + nt * 16 + lm;
            int h = n >> 6, d = n & 63;
            float bi = bia[n];
            #pragma unroll
            for (int r = 0; r < 8; r++) {
                int mrow = m0 + mt * 16 + r + lh * 8;   // C layout: row = VGPR idx (+8 hi half)
                int b = mrow >> 12, s = mrow & (SEQ - 1);
                _Float16 val = f2h(acc[mt][nt][r] + bi);
                if (z == 2)
                    vo[(((size_t)b * NHEAD + h) * HD + d) * SEQ + s] = val;        // V^T
                else {
                    _Float16* dst = (z == 0) ? qo : ko;
                    dst[(((size_t)b * NHEAD + h) * SEQ + s) * HD + d] = val;
                }
            }
        }
    }
}

// ---------------------------------------------------------------------------
// Kernel 3: flash attention. 1 wave = one 16-row q tile; blockDim 128 = 4 tiles.
// grid = (S/16/4, B*H). Streams kseq in chunks of 32: 4 score WMMAs,
// online softmax (shfl_xor row reductions), P transposed via per-wave LDS,
// 4 PV WMMAs into a 16x64 f32 accumulator. ctx written f16 [B,S,D].
// ---------------------------------------------------------------------------
__global__ __launch_bounds__(128) void attn_kernel(const _Float16* __restrict__ q,
                                                   const _Float16* __restrict__ kmat,
                                                   const _Float16* __restrict__ vT,
                                                   const int* __restrict__ mask,
                                                   _Float16* __restrict__ ctx) {
    int lane = threadIdx.x & 31, wave = threadIdx.x >> 5;
    int lm = lane & 15, lh = lane >> 4;
    int bh = blockIdx.y;
    int b = bh / NHEAD, h = bh % NHEAD;
    int q0 = (blockIdx.x * 4 + wave) * 16;

    const _Float16* qp = q    + ((size_t)bh * SEQ + q0) * HD;
    const _Float16* kp = kmat + (size_t)bh * SEQ * HD;
    const _Float16* vp = vT   + (size_t)bh * HD * SEQ;
    const int*      mp = mask + (size_t)b * SEQ;

    __shared__ _Float16 plds[4][16 * 32];
    _Float16* myp = plds[wave];

    FragA aq[2];                               // Q tile as two 16x32 A frags (hd=64)
    {
        const _Float16* qrow = qp + (size_t)lm * HD;
        #pragma unroll
        for (int j = 0; j < 2; j++) {
            aq[j].h[0] = *(const h8*)(qrow + j * 32 + lh * 16);
            aq[j].h[1] = *(const h8*)(qrow + j * 32 + 8 + lh * 16);
        }
    }

    v8f o[4] = {};                             // 16x64 f32 accumulator
    float mrun[8], lrun[8];
    #pragma unroll
    for (int r = 0; r < 8; r++) { mrun[r] = -3.0e38f; lrun[r] = 0.f; }

    for (int kb = 0; kb < SEQ; kb += 32) {
        // ---- scores: two 16x16 tiles over this 32-wide kseq chunk
        v8f st[2];
        #pragma unroll
        for (int t = 0; t < 2; t++) {
            v8f c = {};
            #pragma unroll
            for (int j = 0; j < 2; j++) {
                v16h bk = *(const v16h*)(kp + (size_t)(kb + t * 16 + lm) * HD + j * 32 + lh * 16);
                c = wmma16(aq[j].v, bk, c);
            }
            st[t] = c;
        }
        // ---- scale 1/sqrt(64) + mask
        int mv0 = mp[kb + lm], mv1 = mp[kb + 16 + lm];
        #pragma unroll
        for (int r = 0; r < 8; r++) {
            float s0 = st[0][r] * 0.125f, s1 = st[1][r] * 0.125f;
            st[0][r] = (mv0 == 0) ? NEG_INF : s0;
            st[1][r] = (mv1 == 0) ? NEG_INF : s1;
        }
        // ---- online softmax (row = VGPR idx; reduce across 16-lane half)
        float alpha[8];
        #pragma unroll
        for (int r = 0; r < 8; r++) {
            float mx = fmaxf(st[0][r], st[1][r]);
            mx = fmaxf(mx, __shfl_xor(mx, 1));
            mx = fmaxf(mx, __shfl_xor(mx, 2));
            mx = fmaxf(mx, __shfl_xor(mx, 4));
            mx = fmaxf(mx, __shfl_xor(mx, 8));
            float mnew = fmaxf(mrun[r], mx);
            alpha[r] = __expf(mrun[r] - mnew);
            mrun[r] = mnew;
            float p0 = __expf(st[0][r] - mnew);
            float p1 = __expf(st[1][r] - mnew);
            st[0][r] = p0; st[1][r] = p1;
            float rs = p0 + p1;
            rs += __shfl_xor(rs, 1);
            rs += __shfl_xor(rs, 2);
            rs += __shfl_xor(rs, 4);
            rs += __shfl_xor(rs, 8);
            lrun[r] = lrun[r] * alpha[r] + rs;
        }
        #pragma unroll
        for (int nt = 0; nt < 4; nt++)
            #pragma unroll
            for (int r = 0; r < 8; r++) o[nt][r] *= alpha[r];
        // ---- P: C layout -> A layout through LDS (16x32 f16 per wave)
        #pragma unroll
        for (int r = 0; r < 8; r++) {
            int row = r + lh * 8;
            myp[row * 32 + lm]      = f2h(st[0][r]);
            myp[row * 32 + 16 + lm] = f2h(st[1][r]);
        }
        __syncthreads();
        FragA pa;
        pa.h[0] = *(const h8*)(myp + lm * 32 + lh * 16);
        pa.h[1] = *(const h8*)(myp + lm * 32 + 8 + lh * 16);
        __syncthreads();
        // ---- O += P x V   (V^T layout -> contiguous B-frag loads)
        #pragma unroll
        for (int nt = 0; nt < 4; nt++) {
            v16h vb = *(const v16h*)(vp + (size_t)(nt * 16 + lm) * SEQ + kb + lh * 16);
            o[nt] = wmma16(pa.v, vb, o[nt]);
        }
    }
    // ---- epilogue: divide by row-sum, write ctx[b, s, h*64 + d]
    #pragma unroll
    for (int r = 0; r < 8; r++) {
        float inv = 1.0f / lrun[r];
        int s = q0 + r + lh * 8;
        size_t base = ((size_t)b * SEQ + s) * DMODEL + h * HD;
        #pragma unroll
        for (int nt = 0; nt < 4; nt++)
            ctx[base + nt * 16 + lm] = f2h(o[nt][r] * inv);
    }
}

// ---------------------------------------------------------------------------
// Kernel 4: out = ctx @ Wo^T + bo + residual(x), fp32 output. 32x64 per wave.
// ---------------------------------------------------------------------------
__global__ __launch_bounds__(128) void oproj_kernel(const _Float16* __restrict__ ctx,
                                                    const _Float16* __restrict__ wo,
                                                    const float* __restrict__ bo,
                                                    const float* __restrict__ x,
                                                    float* __restrict__ out) {
    int lane = threadIdx.x & 31, wave = threadIdx.x >> 5;
    int lm = lane & 15, lh = lane >> 4;
    int m0 = blockIdx.x * 128 + wave * 32;
    int n0 = blockIdx.y * 64;
    const _Float16* arow0 = ctx + (size_t)(m0 + lm) * DMODEL;
    const _Float16* arow1 = ctx + (size_t)(m0 + 16 + lm) * DMODEL;
    const _Float16* brow  = wo  + (size_t)(n0 + lm) * DMODEL;
    v8f acc[2][4] = {};
    for (int k = 0; k < DMODEL; k += 32) {
        if (k + 64 < DMODEL) {
            __builtin_prefetch(arow0 + k + 64, 0, 0);
            __builtin_prefetch(arow1 + k + 64, 0, 0);
            __builtin_prefetch(brow + k + 64, 0, 0);
        }
        FragA a0, a1;
        a0.h[0] = *(const h8*)(arow0 + k + lh * 16);
        a0.h[1] = *(const h8*)(arow0 + k + 8 + lh * 16);
        a1.h[0] = *(const h8*)(arow1 + k + lh * 16);
        a1.h[1] = *(const h8*)(arow1 + k + 8 + lh * 16);
        #pragma unroll
        for (int nt = 0; nt < 4; nt++) {
            v16h bm = *(const v16h*)(wo + (size_t)(n0 + nt * 16 + lm) * DMODEL + k + lh * 16);
            acc[0][nt] = wmma16(a0.v, bm, acc[0][nt]);
            acc[1][nt] = wmma16(a1.v, bm, acc[1][nt]);
        }
    }
    #pragma unroll
    for (int mt = 0; mt < 2; mt++) {
        #pragma unroll
        for (int nt = 0; nt < 4; nt++) {
            int n = n0 + nt * 16 + lm;
            float bi = bo[n];
            #pragma unroll
            for (int r = 0; r < 8; r++) {
                size_t idx = (size_t)(m0 + mt * 16 + r + lh * 8) * DMODEL + n;
                out[idx] = acc[mt][nt][r] + bi + x[idx];
            }
        }
    }
}

// ---------------------------------------------------------------------------
// Launch: inputs in setup_inputs() order.
// Workspace (f16 elems): [4*D*D weights][xn M*D][Q][K][V^T][ctx] ~= 68 MB.
// ---------------------------------------------------------------------------
extern "C" void kernel_launch(void* const* d_in, const int* in_sizes, int n_in,
                              void* d_out, int out_size, void* d_ws, size_t ws_size,
                              hipStream_t stream) {
    const float* x     = (const float*)d_in[0];
    const int*   mask  = (const int*)d_in[1];
    const float* gamma = (const float*)d_in[2];
    const float* beta  = (const float*)d_in[3];
    const float* Wq    = (const float*)d_in[4];
    const float* bq    = (const float*)d_in[5];
    const float* Wk    = (const float*)d_in[6];
    const float* bk    = (const float*)d_in[7];
    const float* Wv    = (const float*)d_in[8];
    const float* bv    = (const float*)d_in[9];
    const float* Wo    = (const float*)d_in[10];
    const float* bo    = (const float*)d_in[11];
    float* out = (float*)d_out;

    const size_t DD = (size_t)DMODEL * DMODEL;   // 589824
    const size_t MD = (size_t)MTOT * DMODEL;     // 6291456
    _Float16* ws16 = (_Float16*)d_ws;
    _Float16* whalf = ws16;                      // [4][D*D]  (q,k,v,o)
    _Float16* xn    = ws16 + 4 * DD;
    _Float16* qb    = xn + MD;                   // [B,H,S,HD]
    _Float16* kb16  = qb + MD;                   // [B,H,S,HD]
    _Float16* vbT   = kb16 + MD;                 // [B,H,HD,S]
    _Float16* ctx   = vbT + MD;                  // [B,S,D]
    (void)in_sizes; (void)n_in; (void)out_size; (void)ws_size;

    cvt_w_kernel<<<dim3((unsigned)(DD / 256), 4), 256, 0, stream>>>(Wq, Wk, Wv, Wo, whalf);
    ln_kernel<<<MTOT, 256, 0, stream>>>(x, gamma, beta, xn);
    qkv_kernel<<<dim3(MTOT / 128, DMODEL / 64, 3), 128, 0, stream>>>(
        xn, whalf, bq, bk, bv, qb, kb16, vbT);
    attn_kernel<<<dim3(SEQ / 64, NBATCH * NHEAD), 128, 0, stream>>>(
        qb, kb16, vbT, mask, ctx);
    oproj_kernel<<<dim3(MTOT / 128, DMODEL / 64), 128, 0, stream>>>(
        ctx, whalf + 3 * DD, bo, x, out);
}
